// SVTRBlock_53437983097259
// MI455X (gfx1250) — compile-verified, matrix operand order checked by reading, not run on hardware
//
#include <hip/hip_runtime.h>
#include <hip/hip_bf16.h>

typedef _Float16 f16;
typedef _Float16 half4 __attribute__((ext_vector_type(4)));
typedef _Float16 half8 __attribute__((ext_vector_type(8)));
typedef _Float16 v16h  __attribute__((ext_vector_type(16)));
typedef float    v4f   __attribute__((ext_vector_type(4)));
typedef float    v8f   __attribute__((ext_vector_type(8)));
typedef int      v4i   __attribute__((vector_size(16)));   // matches builtin proto

#define DEV __device__ __forceinline__

// ---- CDNA5 async global->LDS copy (ASYNCcnt path), with safe fallback ------
#if defined(__has_builtin)
#if __has_builtin(__builtin_amdgcn_global_load_async_to_lds_b128)
#define HAVE_ASYNC_LDS 1
#endif
#endif

#ifdef HAVE_ASYNC_LDS
typedef __attribute__((address_space(1))) v4i* as1_v4i;
typedef __attribute__((address_space(3))) v4i* as3_v4i;
#endif

DEV void copy16_g2l(const f16* g, f16* l) {
#ifdef HAVE_ASYNC_LDS
  __builtin_amdgcn_global_load_async_to_lds_b128((as1_v4i)(void*)g,
                                                 (as3_v4i)(void*)l, 0, 0);
#else
  *(half8*)l = *(const half8*)g;
#endif
}

DEV void wait_async_le4() {
#ifdef HAVE_ASYNC_LDS
#if __has_builtin(__builtin_amdgcn_s_wait_asynccnt)
  __builtin_amdgcn_s_wait_asynccnt(4);
#else
  asm volatile("s_wait_asynccnt 0x4" ::: "memory");
#endif
#endif
}

DEV void wait_async_zero() {
#ifdef HAVE_ASYNC_LDS
#if __has_builtin(__builtin_amdgcn_s_wait_asynccnt)
  __builtin_amdgcn_s_wait_asynccnt(0);
#else
  asm volatile("s_wait_asynccnt 0x0" ::: "memory");
#endif
#endif
}

// ---- WMMA helpers ----------------------------------------------------------
union H16u { half8 h8[2]; v16h v; };

// Load a 16x32 f16 fragment (A layout; also valid for B when memory holds B^T
// row-major) with row stride `stride` (halves). Per ISA 7.12.2: lanes 0-15
// row=lane; K halves {sel*8+0..7, sel*8+16..23}, sel = lane>=16 -> two
// contiguous 16B loads per lane.
DEV v16h load_frag(const f16* base, int stride, int lane) {
  const int r   = lane & 15;
  const int sel = (lane >> 4) & 1;
  const f16* p0 = base + r * stride + sel * 8;
  H16u u;
  u.h8[0] = *(const half8*)(p0);
  u.h8[1] = *(const half8*)(p0 + 16);
  return u.v;
}

DEV v8f wmma16x16x32(v16h a, v16h b, v8f c) {
  return __builtin_amdgcn_wmma_f32_16x16x32_f16(false, a, false, b, (short)0, c,
                                                false, false);
}

// ---- constants -------------------------------------------------------------
static constexpr int BATCH = 32, NTOK = 1024, DIM = 256, NH = 8, HD = 32;
static constexpr int MTOT = BATCH * NTOK;               // 32768 rows
static constexpr float QSCALE = 0.17677669529663687f;   // 32^-0.5

// ---- prep kernels ----------------------------------------------------------
__global__ void conv_f32_f16(const float* __restrict__ s, f16* __restrict__ d,
                             int n4) {
  int i = blockIdx.x * 256 + threadIdx.x;
  if (i < n4) {
    v4f v = *(const v4f*)(s + (size_t)i * 4);
    half4 h = {(f16)v[0], (f16)v[1], (f16)v[2], (f16)v[3]};
    *(half4*)(d + (size_t)i * 4) = h;
  }
}

// src[K][N] f32 -> dst[N][K] f16 (weights become B^T, row-major over K)
__global__ void transpose_f32_f16(const float* __restrict__ s,
                                  f16* __restrict__ d, int K, int N) {
  int i = blockIdx.x * 256 + threadIdx.x;
  if (i < K * N) {
    int k = i / N, n = i - k * N;
    d[(size_t)n * K + k] = (f16)s[i];
  }
}

// ---- WMMA GEMM -------------------------------------------------------------
// C[M x N] = A[M x K] * B[K x N] (+bias), A row-major f16, Bt = B^T row-major.
// Block tile 128x128, K step 32, double-buffered async LDS staging,
// 8 waves (2x4), 4x2 fragments per wave.
// EPI: 0 = QKV split (scale q, k row-major, v transposed)
//      1 = bias -> f32 out
//      2 = bias + exact GELU -> f16 out
template <int EPI>
__global__ __launch_bounds__(256) void gemm_f16(
    const f16* __restrict__ A, const f16* __restrict__ Bt,
    const float* __restrict__ bias, float* __restrict__ outF,
    f16* __restrict__ outH, f16* __restrict__ qb, f16* __restrict__ kbuf,
    f16* __restrict__ vtb, int K, int Nld) {
  constexpr int LDT = 40;  // padded LDS row stride (halves), keeps 16B align
  __shared__ __align__(16) f16 As[2][128 * LDT];
  __shared__ __align__(16) f16 Bs[2][128 * LDT];
  const int tid  = threadIdx.x;
  const int lane = tid & 31;
  const int wave = tid >> 5;
  const int wm   = wave >> 2;  // 0..1
  const int wn   = wave & 3;   // 0..3
  const int m0   = blockIdx.y * 128;
  const int n0   = blockIdx.x * 128;

  const v8f zero = {0.f, 0.f, 0.f, 0.f, 0.f, 0.f, 0.f, 0.f};
  v8f acc[4][2];
  for (int i = 0; i < 4; ++i)
    for (int j = 0; j < 2; ++j) acc[i][j] = zero;

  // This thread's 32B chunk of the 128x32 tile (2 threads per row).
  const int row = tid >> 1;
  const int cb  = (tid & 1) * 16;
  const f16* ag = A  + (size_t)(m0 + row) * K + cb;
  const f16* bg = Bt + (size_t)(n0 + row) * K + cb;

  auto stage = [&](int buf, int k0) {
    f16* la = As[buf] + row * LDT + cb;
    f16* lb = Bs[buf] + row * LDT + cb;
    copy16_g2l(ag + k0,     la);
    copy16_g2l(ag + k0 + 8, la + 8);
    copy16_g2l(bg + k0,     lb);
    copy16_g2l(bg + k0 + 8, lb + 8);
  };

  const int nsteps = K >> 5;
  stage(0, 0);
  for (int s = 0; s < nsteps; ++s) {
    const int cur = s & 1;
    const bool more = (s + 1) < nsteps;
    if (more) stage(1 - cur, (s + 1) << 5);     // prefetch next tile (async)
    if (more) wait_async_le4();                 // wait only for current tile
    else      wait_async_zero();
    __syncthreads();
    v16h aF[4], bF[2];
    for (int i = 0; i < 4; ++i)
      aF[i] = load_frag(As[cur] + (wm * 64 + i * 16) * LDT, LDT, lane);
    for (int j = 0; j < 2; ++j)
      bF[j] = load_frag(Bs[cur] + (wn * 32 + j * 16) * LDT, LDT, lane);
    for (int i = 0; i < 4; ++i)
      for (int j = 0; j < 2; ++j)
        acc[i][j] = wmma16x16x32(aF[i], bF[j], acc[i][j]);
    __syncthreads();  // protect buffer cur before next prefetch overwrites
  }

  // Epilogue: C frag layout — VGPR v holds M = v + 8*(lane>=16), N = lane&15
  const int hi = (lane >> 4) & 1;
  const int nc = lane & 15;
  for (int i = 0; i < 4; ++i)
    for (int j = 0; j < 2; ++j) {
      const int gmB = m0 + wm * 64 + i * 16 + 8 * hi;
      const int gn  = n0 + wn * 32 + j * 16 + nc;
      const float bval = bias[gn];
      for (int v = 0; v < 8; ++v) {
        const int gm = gmB + v;
        float val = acc[i][j][v] + bval;
        if constexpr (EPI == 0) {
          const int which = gn >> 8;          // 0=q 1=k 2=v
          const int c = gn & 255;
          const int head = c >> 5, d = c & 31;
          const int bidx = gm >> 10, tok = gm & 1023;
          const size_t bh = (size_t)bidx * 8 + head;
          if (which == 0)
            qb[(bh * 1024 + tok) * 32 + d] = (f16)(val * QSCALE);
          else if (which == 1)
            kbuf[(bh * 1024 + tok) * 32 + d] = (f16)val;
          else
            vtb[(bh * 32 + d) * 1024 + tok] = (f16)val;  // transposed V
        } else if constexpr (EPI == 1) {
          outF[(size_t)gm * Nld + gn] = val;
        } else {
          const float gl = 0.5f * val * (1.0f + erff(val * 0.70710678118654752f));
          outH[(size_t)gm * Nld + gn] = (f16)gl;
        }
      }
    }
}

// ---- local attention -------------------------------------------------------
// One wave per 16-query tile. 7x11 window => key tiles: 7 h-rows x 3 w-tiles.
__global__ __launch_bounds__(32) void attn_local(
    const f16* __restrict__ qb, const f16* __restrict__ kb,
    const f16* __restrict__ vtb, f16* __restrict__ ctx) {
  __shared__ float sc[16][336];                 // raw scores (<=21 tiles * 16)
  __shared__ __align__(16) f16 at[16][352];     // normalized attn, padded
  __shared__ int kbs[24];

  const int lane  = threadIdx.x;
  const int bh    = blockIdx.x >> 6;            // batch*8 + head
  const int qtile = blockIdx.x & 63;
  const int qt0   = qtile * 16;
  const int qh    = qtile >> 2;                 // grid is 16 x 64, 4 w-tiles
  const int qwt   = qtile & 3;
  const int qw0   = qwt * 16;

  // Enumerate reachable key tiles (uniform across the wave).
  int ntiles = 0;
  for (int kh = qh - 3; kh <= qh + 3; ++kh) {
    if ((unsigned)kh > 15u) continue;
    for (int kwt = qwt - 1; kwt <= qwt + 1; ++kwt) {
      if ((unsigned)kwt > 3u) continue;
      if (lane == 0) kbs[ntiles] = (kh * 4 + kwt) * 16;
      ++ntiles;
    }
  }
  const int padtiles = (ntiles + 1) & ~1;
  if (lane == 0)
    for (int t = ntiles; t < padtiles; ++t) kbs[t] = 0;
  __syncthreads();

  const v8f zero = {0.f, 0.f, 0.f, 0.f, 0.f, 0.f, 0.f, 0.f};
  const int hi = (lane >> 4) & 1;
  const int nc = lane & 15;

  // Pass 1: scores = (q*scale) @ k^T per key tile (K = HEAD_DIM = 32).
  const v16h qF = load_frag(qb + ((size_t)bh * 1024 + qt0) * 32, 32, lane);
  for (int t = 0; t < ntiles; ++t) {
    const int kt0 = kbs[t];
    const v16h kF = load_frag(kb + ((size_t)bh * 1024 + kt0) * 32, 32, lane);
    v8f s = wmma16x16x32(qF, kF, zero);
    const int ktok = kt0 + nc;
    const int kh2 = ktok >> 6, kw2 = ktok & 63;
    int dh = qh - kh2; if (dh < 0) dh = -dh;
    for (int v = 0; v < 8; ++v) {
      const int qw = qw0 + v + 8 * hi;
      int dw = qw - kw2; if (dw < 0) dw = -dw;
      const bool ok = (dh <= 3) && (dw <= 5);
      sc[v + 8 * hi][t * 16 + nc] = ok ? s[v] : -1e30f;
    }
  }
  __syncthreads();

  // Pass 2: row-wise softmax, lanes 0..15 own one row each.
  if (lane < 16) {
    const int r = lane;
    const int ncols = ntiles * 16;
    float mx = -1e30f;
    for (int j = 0; j < ncols; ++j) mx = fmaxf(mx, sc[r][j]);
    float sum = 0.f;
    for (int j = 0; j < ncols; ++j) {
      const float e = __expf(sc[r][j] - mx);
      sc[r][j] = e;
      sum += e;
    }
    const float inv = 1.0f / sum;
    for (int j = 0; j < ncols; ++j) at[r][j] = (f16)(sc[r][j] * inv);
    for (int j = ncols; j < padtiles * 16; ++j) at[r][j] = (f16)0.f;
  }
  __syncthreads();

  // Pass 3: ctx = attn @ V, key dim consumed 32 at a time (two tiles).
  v8f acc0 = zero, acc1 = zero;
  const f16* vbase = vtb + (size_t)bh * 32 * 1024;
  const f16* vr0 = vbase + (size_t)(nc) * 1024;        // d = 0..15
  const f16* vr1 = vbase + (size_t)(16 + nc) * 1024;   // d = 16..31
  const int o = hi * 8;
  for (int j = 0; j < padtiles / 2; ++j) {
    const v16h aF = load_frag(&at[0][0] + j * 32, 352, lane);
    const int ktA = kbs[2 * j], ktB = kbs[2 * j + 1];
    H16u u0, u1;
    u0.h8[0] = *(const half8*)(vr0 + ktA + o);
    u0.h8[1] = *(const half8*)(vr0 + ktB + o);
    u1.h8[0] = *(const half8*)(vr1 + ktA + o);
    u1.h8[1] = *(const half8*)(vr1 + ktB + o);
    acc0 = wmma16x16x32(aF, u0.v, acc0);
    acc1 = wmma16x16x32(aF, u1.v, acc1);
  }

  const int b = bh >> 3, head = bh & 7;
  for (int v = 0; v < 8; ++v) {
    const int tok = qt0 + v + 8 * hi;
    const size_t rowb = ((size_t)b * 1024 + tok) * 256 + head * 32;
    ctx[rowb + nc]      = (f16)acc0[v];
    ctx[rowb + 16 + nc] = (f16)acc1[v];
  }
}

// ---- residual + layernorm (C = 256), vectorized ---------------------------
__global__ __launch_bounds__(256) void ln_residual(
    const float* __restrict__ xin, const float* __restrict__ yin,
    const float* __restrict__ g, const float* __restrict__ bta,
    float* __restrict__ outF, f16* __restrict__ outH) {
  const int lane = threadIdx.x & 31;
  const int wave = threadIdx.x >> 5;
  const size_t t = (size_t)blockIdx.x * 8 + wave;
  const size_t base = t * 256 + (size_t)lane * 8;   // 8 contiguous floats/lane

  v4f va = *(const v4f*)(xin + base)     + *(const v4f*)(yin + base);
  v4f vb = *(const v4f*)(xin + base + 4) + *(const v4f*)(yin + base + 4);
  float s = va[0] + va[1] + va[2] + va[3] + vb[0] + vb[1] + vb[2] + vb[3];
  for (int m = 16; m; m >>= 1) s += __shfl_xor(s, m, 32);
  const float mu = s * (1.0f / 256.0f);
  float ss = 0.f;
  for (int j = 0; j < 4; ++j) { float d0 = va[j] - mu; ss += d0 * d0; }
  for (int j = 0; j < 4; ++j) { float d0 = vb[j] - mu; ss += d0 * d0; }
  for (int m = 16; m; m >>= 1) ss += __shfl_xor(ss, m, 32);
  const float rstd = rsqrtf(ss * (1.0f / 256.0f) + 1e-6f);

  const int c0 = lane * 8;
  v4f ga = *(const v4f*)(g + c0),   gb = *(const v4f*)(g + c0 + 4);
  v4f ba = *(const v4f*)(bta + c0), bb = *(const v4f*)(bta + c0 + 4);
  v4f oa, ob;
  for (int j = 0; j < 4; ++j) oa[j] = (va[j] - mu) * rstd * ga[j] + ba[j];
  for (int j = 0; j < 4; ++j) ob[j] = (vb[j] - mu) * rstd * gb[j] + bb[j];
  *(v4f*)(outF + base)     = oa;
  *(v4f*)(outF + base + 4) = ob;
  if (outH) {
    half4 ha = {(f16)oa[0], (f16)oa[1], (f16)oa[2], (f16)oa[3]};
    half4 hb = {(f16)ob[0], (f16)ob[1], (f16)ob[2], (f16)ob[3]};
    *(half4*)(outH + base)     = ha;
    *(half4*)(outH + base + 4) = hb;
  }
}

// ---- launch ----------------------------------------------------------------
extern "C" void kernel_launch(void* const* d_in, const int* in_sizes, int n_in,
                              void* d_out, int out_size, void* d_ws,
                              size_t ws_size, hipStream_t stream) {
  (void)in_sizes; (void)n_in; (void)out_size; (void)ws_size;
  const float* x      = (const float*)d_in[0];
  const float* qkv_w  = (const float*)d_in[1];
  const float* qkv_b  = (const float*)d_in[2];
  const float* proj_w = (const float*)d_in[3];
  const float* proj_b = (const float*)d_in[4];
  const float* fc1_w  = (const float*)d_in[5];
  const float* fc1_b  = (const float*)d_in[6];
  const float* fc2_w  = (const float*)d_in[7];
  const float* fc2_b  = (const float*)d_in[8];
  const float* ln1_g  = (const float*)d_in[9];
  const float* ln1_b  = (const float*)d_in[10];
  const float* ln2_g  = (const float*)d_in[11];
  const float* ln2_b  = (const float*)d_in[12];

  char* w = (char*)d_ws;
  constexpr size_t SZ_XH    = (size_t)MTOT * DIM * 2;                 // 16 MB
  constexpr size_t SZ_WQKV  = (size_t)768 * 256 * 2;
  constexpr size_t SZ_WPRJ  = (size_t)256 * 256 * 2;
  constexpr size_t SZ_WFC1  = (size_t)1024 * 256 * 2;
  constexpr size_t SZ_WFC2  = (size_t)256 * 1024 * 2;
  constexpr size_t SZ_HEADB = (size_t)BATCH * NH * NTOK * HD * 2;     // 16 MB
  constexpr size_t SZ_F32   = (size_t)MTOT * DIM * 4;                 // 32 MB

  size_t off = 0;
  f16* xh     = (f16*)(w + off); off += SZ_XH;
  f16* wqkvT  = (f16*)(w + off); off += SZ_WQKV;
  f16* wprojT = (f16*)(w + off); off += SZ_WPRJ;
  f16* wfc1T  = (f16*)(w + off); off += SZ_WFC1;
  f16* wfc2T  = (f16*)(w + off); off += SZ_WFC2;
  size_t attn_base = off;
  f16* qbuf   = (f16*)(w + off); off += SZ_HEADB;
  f16* kbuf   = (f16*)(w + off); off += SZ_HEADB;
  f16* vtbuf  = (f16*)(w + off); off += SZ_HEADB;
  f16* ctx    = (f16*)(w + off); off += SZ_HEADB;
  float* y1f  = (float*)(w + off); off += SZ_F32;
  float* x1f  = (float*)(w + off); off += SZ_F32;
  // Aliases (lifetimes disjoint): x1h over xh, hbuf over q/k/vT/ctx, y2 over y1
  f16*   x1h  = xh;
  f16*   hbuf = (f16*)(w + attn_base);                                // 64 MB
  float* y2f  = y1f;

  // 1) f16 copies: activations + transposed weights
  conv_f32_f16<<<(MTOT * DIM / 4 + 255) / 256, 256, 0, stream>>>(x, xh,
                                                                 MTOT * DIM / 4);
  transpose_f32_f16<<<(256 * 768 + 255) / 256, 256, 0, stream>>>(qkv_w, wqkvT, 256, 768);
  transpose_f32_f16<<<(256 * 256 + 255) / 256, 256, 0, stream>>>(proj_w, wprojT, 256, 256);
  transpose_f32_f16<<<(256 * 1024 + 255) / 256, 256, 0, stream>>>(fc1_w, wfc1T, 256, 1024);
  transpose_f32_f16<<<(1024 * 256 + 255) / 256, 256, 0, stream>>>(fc2_w, wfc2T, 1024, 256);

  // 2) QKV projection (M=32768, K=256, N=768) with head-split epilogue
  gemm_f16<0><<<dim3(768 / 128, MTOT / 128), 256, 0, stream>>>(
      xh, wqkvT, qkv_b, nullptr, nullptr, qbuf, kbuf, vtbuf, 256, 768);

  // 3) masked local attention (one wave per 16-query tile)
  attn_local<<<BATCH * NH * (NTOK / 16), 32, 0, stream>>>(qbuf, kbuf, vtbuf, ctx);

  // 4) output projection -> y1 (f32)
  gemm_f16<1><<<dim3(256 / 128, MTOT / 128), 256, 0, stream>>>(
      ctx, wprojT, proj_b, y1f, nullptr, nullptr, nullptr, nullptr, 256, 256);

  // 5) LN1(x + y1) -> x1f (f32) and x1h (f16)
  ln_residual<<<MTOT / 8, 256, 0, stream>>>(x, y1f, ln1_g, ln1_b, x1f, x1h);

  // 6) fc1 + exact GELU -> h (f16), (M=32768, K=256, N=1024)
  gemm_f16<2><<<dim3(1024 / 128, MTOT / 128), 256, 0, stream>>>(
      x1h, wfc1T, fc1_b, nullptr, hbuf, nullptr, nullptr, nullptr, 256, 1024);

  // 7) fc2 -> y2 (f32), (M=32768, K=1024, N=256)
  gemm_f16<1><<<dim3(256 / 128, MTOT / 128), 256, 0, stream>>>(
      hbuf, wfc2T, fc2_b, y2f, nullptr, nullptr, nullptr, nullptr, 1024, 256);

  // 8) LN2(x1 + y2) -> out (f32)
  ln_residual<<<MTOT / 8, 256, 0, stream>>>(x1f, y2f, ln2_g, ln2_b,
                                            (float*)d_out, nullptr);
}
